// NeuralHawkes_79018808312176
// MI455X (gfx1250) — compile-verified, matrix operand order checked by reading, not run on hardware
//
#include <hip/hip_runtime.h>
#include <hip/hip_bf16.h>
#include <math.h>

// ---------------- problem constants ----------------
#define RR 128          // B*M rows
#define T2 256
#define TS 255          // scan steps
#define SS 256          // samples per (b,m)
#define HD 512          // hidden
#define KK 50           // event types scored
#define NE 53           // K+3 distinct event codes
#define NG 3584         // 7*H gate columns
#define NWG 8           // persistent workgroups in recurrence
#define EPSV 2.220446049250313e-16f

typedef __attribute__((ext_vector_type(8)))  float  v8f;
typedef __attribute__((ext_vector_type(16))) __bf16 v16bf;

// ---------------- helpers ----------------
__device__ __forceinline__ unsigned short f2bf(float f) {
    union { float f; unsigned u; } v; v.f = f;
    unsigned u = v.u;
    unsigned r = (u + 0x7FFFu + ((u >> 16) & 1u)) >> 16;
    return (unsigned short)r;
}
__device__ __forceinline__ float bf2f(unsigned short b) {
    union { unsigned u; float f; } v; v.u = ((unsigned)b) << 16;
    return v.f;
}
__device__ __forceinline__ __bf16 u2bf(unsigned short u) {
    union { unsigned short u; __bf16 h; } c; c.u = u; return c.h;
}
__device__ __forceinline__ float sigm(float x)  { return 1.f / (1.f + __expf(-x)); }
__device__ __forceinline__ float splus(float x) {
    return x > 0.f ? x + log1pf(__expf(-x)) : log1pf(__expf(x));
}

// A/B fragment: lanes 0-15 hold k {0..7,16..23}, lanes 16-31 hold {8..15,24..31}
__device__ __forceinline__ v16bf load_frag(const unsigned short* base, int half) {
    v16bf r;
    const unsigned short* p0 = base + 8 * half;
    const unsigned short* p1 = base + 16 + 8 * half;
#pragma unroll
    for (int e = 0; e < 8; ++e) { r[e] = u2bf(p0[e]); r[8 + e] = u2bf(p1[e]); }
    return r;
}
__device__ __forceinline__ v8f wmma_bf16(v16bf a, v16bf b, v8f c) {
    return __builtin_amdgcn_wmma_f32_16x16x32_bf16(false, a, false, b, (short)0, c,
                                                   false, false);
}

// ---------------- workspace layout (all offsets multiples of 256B) ----------------
#define SZ_P    (NE * NG * 4)              // f32 gate table P[53][3584]
#define SZ_WRT  (NG * HD * 2)              // bf16 W_rnn recurrent, transposed [3584][512]
#define SZ_WLAM (64 * HD * 2)              // bf16 W_lambda padded [64][512]
#define SZ_BIG  (RR * TS * HD * 2)         // bf16 per-timestep logs [32640][512]
#define SZ_ST   (RR * HD * 4)              // f32 carry state
#define SZ_LLT  (RR * TS * 4)
#define OFF_P    ((size_t)0)
#define OFF_WRT  (OFF_P    + SZ_P)
#define OFF_WLAM (OFF_WRT  + SZ_WRT)
#define OFF_HID  (OFF_WLAM + SZ_WLAM)
#define OFF_CEL  (OFF_HID  + SZ_BIG)
#define OFF_CBA  (OFF_CEL  + SZ_BIG)
#define OFF_GD   (OFF_CBA  + SZ_BIG)
#define OFF_GO   (OFF_GD   + SZ_BIG)
#define OFF_CST  (OFF_GO   + SZ_BIG)
#define OFF_CBS  (OFF_CST  + SZ_ST)
#define OFF_LLT  (OFF_CBS  + SZ_ST)
#define OFF_NUM  (OFF_LLT  + SZ_LLT)
#define OFF_DEN  (OFF_NUM  + 512)
#define OFF_CTR  (OFF_DEN  + 512)

// ---------------- prep kernels ----------------
__global__ void k_prep_p(const float* __restrict__ Emb, const float* __restrict__ W_rnn,
                         const float* __restrict__ b_rnn, float* __restrict__ P) {
    int id = blockIdx.x * blockDim.x + threadIdx.x;
    if (id >= NE * NG) return;
    int k = id / NG, n = id % NG;
    float s = b_rnn[n];
    const float* er = Emb + (size_t)k * HD;
    for (int h = 0; h < HD; ++h) s += er[h] * W_rnn[(size_t)h * NG + n];
    P[id] = s;
}

__global__ void k_prep_w(const float* __restrict__ W_rnn, unsigned short* __restrict__ WrT) {
    int id = blockIdx.x * blockDim.x + threadIdx.x;
    if (id >= NG * HD) return;
    int n = id >> 9, h = id & 511;                     // WrT[n][h] = W_rnn[512+h][n]
    WrT[id] = f2bf(W_rnn[(size_t)(HD + h) * NG + n]);
}

__global__ void k_prep_misc(const float* __restrict__ W_lambda, unsigned short* __restrict__ Wlam,
                            float* __restrict__ c_state, float* __restrict__ cb_state,
                            float* __restrict__ num, float* __restrict__ den,
                            unsigned* __restrict__ ctr) {
    int id = blockIdx.x * blockDim.x + threadIdx.x;
    if (id < 64 * HD) Wlam[id] = (id < KK * HD) ? f2bf(W_lambda[id]) : (unsigned short)0;
    if (id < RR * HD) { c_state[id] = 0.f; cb_state[id] = 0.f; }
    if (id < RR) { num[id] = 0.f; den[id] = 0.f; }
    if (id == 0) *ctr = 0u;
}

// ---------------- persistent recurrence kernel ----------------
// 8 WGs x 512 threads (16 waves). WG b owns hidden cols [64b, 64b+64).
// wave w: row-tile (w & 7), hidden sub-tiles {2*(w>>3), 2*(w>>3)+1}.
__global__ void __launch_bounds__(512)
k_recurrence(const float* __restrict__ P, const unsigned short* __restrict__ WrT,
             unsigned short* __restrict__ hidden,
             unsigned short* __restrict__ cellA, unsigned short* __restrict__ cbarA,
             unsigned short* __restrict__ gdA, unsigned short* __restrict__ goA,
             float* __restrict__ c_state, float* __restrict__ cb_state,
             const int* __restrict__ event, const float* __restrict__ dtime,
             unsigned* __restrict__ ctr) {
    const int tid  = threadIdx.x;
    const int wave = tid >> 5, lane = tid & 31;
    const int half = lane >> 4, l16 = lane & 15;
    const int rbase = (wave & 7) * 16;
    const int arow  = rbase + l16;                 // A-fragment row (bm)
    const int hs0   = (wave >> 3) * 2;             // first hidden sub-tile
    const int jwg   = blockIdx.x * 64;

    for (int t = 0; t < TS; ++t) {
        if (t) {  // grid barrier: previous step's hidden fully stored
            __syncthreads();
            if (tid == 0) {
                __threadfence();
                atomicAdd(ctr, 1u);
                const unsigned tgt = (unsigned)NWG * (unsigned)t;
                while (atomicAdd(ctr, 0u) < tgt) { }
                __threadfence();
            }
            __syncthreads();
        }
        const unsigned short* hrowA =
            hidden + ((size_t)arow * TS + (t - 1)) * HD;   // only used when t>0

        // per-element row constants (shared by both hidden sub-tiles)
        const float* prow_[8];
        float dt_[8];
        int rowi_[8];
#pragma unroll
        for (int i = 0; i < 8; ++i) {
            const int rowi = rbase + i + 8 * half;
            rowi_[i] = rowi;
            prow_[i] = P + (size_t)event[rowi * T2 + t] * NG;
            dt_[i]   = dtime[rowi * T2 + t + 1];
        }

        for (int hs = hs0; hs < hs0 + 2; ++hs) {
            const int jcol = jwg + hs * 16 + l16;

            v8f acc[7];
            // init accumulators from precomputed input table P[event]
#pragma unroll
            for (int i = 0; i < 8; ++i) {
#pragma unroll
                for (int g = 0; g < 7; ++g) acc[g][i] = prow_[i][g * HD + jcol];
            }
            // recurrent GEMM: acc += h_{t-1} @ Wr (bf16 WMMA), 2-deep pipeline
            if (t) {
                const unsigned short* wr[7];
#pragma unroll
                for (int g = 0; g < 7; ++g)
                    wr[g] = WrT + (size_t)((g << 9) + jcol) * HD;

                v16bf abuf[2];
                v16bf bbuf[2][7];
                abuf[0] = load_frag(hrowA, half);
#pragma unroll
                for (int g = 0; g < 7; ++g) bbuf[0][g] = load_frag(wr[g], half);

                for (int kk = 0; kk < HD; kk += 64) {
                    // stage buffer 1 from kk+32 while consuming buffer 0
                    abuf[1] = load_frag(hrowA + kk + 32, half);
#pragma unroll
                    for (int g = 0; g < 7; ++g)
                        bbuf[1][g] = load_frag(wr[g] + kk + 32, half);
#pragma unroll
                    for (int g = 0; g < 7; ++g)
                        acc[g] = wmma_bf16(abuf[0], bbuf[0][g], acc[g]);
                    // stage buffer 0 from kk+64 while consuming buffer 1
                    if (kk + 64 < HD) {
                        abuf[0] = load_frag(hrowA + kk + 64, half);
#pragma unroll
                        for (int g = 0; g < 7; ++g)
                            bbuf[0][g] = load_frag(wr[g] + kk + 64, half);
                    }
#pragma unroll
                    for (int g = 0; g < 7; ++g)
                        acc[g] = wmma_bf16(abuf[1], bbuf[1][g], acc[g]);
                }
            }
            // gates + CT-LSTM update; each lane-element owns one (bm, j)
#pragma unroll
            for (int i = 0; i < 8; ++i) {
                const int rowi = rowi_[i];
                const float dt = dt_[i];
                const float gi  = sigm(acc[0][i]);
                const float gf  = sigm(acc[1][i]);
                const float go_ = sigm(acc[2][i]);
                const float gz  = tanhf(acc[3][i]);
                const float gib = sigm(acc[4][i]);
                const float gfb = sigm(acc[5][i]);
                const float gd  = splus(acc[6][i]);
                const size_t sidx = (size_t)rowi * HD + jcol;
                const float c_o  = c_state[sidx];
                const float cb_o = cb_state[sidx];
                const float cell = gf * c_o + gi * gz;
                const float cbn  = gfb * cb_o + gib * gz;
                const float cm   = cbn + (cell - cbn) * __expf(-gd * dt);
                const float hm   = go_ * tanhf(cm);
                c_state[sidx]  = cm;     // carry is the decayed cell
                cb_state[sidx] = cbn;
                const size_t o = ((size_t)rowi * TS + t) * HD + jcol;
                cellA[o] = f2bf(cell);
                cbarA[o] = f2bf(cbn);
                gdA[o]   = f2bf(gd);
                goA[o]   = f2bf(go_);
                hidden[o] = f2bf(hm);
            }
        }
    }
}

// ---------------- llt: one wave per (bm, t) ----------------
__global__ void k_llt(const unsigned short* __restrict__ hidden,
                      const float* __restrict__ W_lambda, const int* __restrict__ event,
                      float* __restrict__ llt) {
    const int gw = blockIdx.x * (blockDim.x >> 5) + (threadIdx.x >> 5);
    const int lane = threadIdx.x & 31;
    const int bm = gw / TS, t = gw % TS;
    const int traw = event[bm * T2 + t + 1];
    const int tgt = traw < KK ? traw : 0;
    const unsigned short* hrow = hidden + ((size_t)bm * TS + t) * HD;
    const float* wrow = W_lambda + (size_t)tgt * HD;
    float acc = 0.f;
#pragma unroll
    for (int ii = 0; ii < 16; ++ii) {
        const int h = ii * 32 + lane;
        acc += bf2f(hrow[h]) * wrow[h];
    }
    for (int off = 16; off; off >>= 1) acc += __shfl_down(acc, off, 32);
    if (!lane) llt[gw] = logf(splus(acc) + EPSV);
}

// ---------------- sampling integral: fused gather + WMMA + softplus-sum ----------------
#define SWV 2   // waves per block
__global__ void k_sample(const unsigned short* __restrict__ cellA,
                         const unsigned short* __restrict__ cbarA,
                         const unsigned short* __restrict__ gdA,
                         const unsigned short* __restrict__ goA,
                         const unsigned short* __restrict__ Wlam,
                         const int* __restrict__ idxs, const float* __restrict__ dts,
                         const float* __restrict__ msk,
                         float* __restrict__ num, float* __restrict__ den) {
    __shared__ unsigned short hs[SWV][16][HD];
    const int wv = threadIdx.x >> 5, lane = threadIdx.x & 31;
    const int half = lane >> 4, l16 = lane & 15;
    const int gw = blockIdx.x * SWV + wv;
    const int sb = gw * 16;

    // phase 1: reconstruct 16 sampled hiddens into LDS (bf16)
    for (int si = 0; si < 16; ++si) {
        const int spl = sb + si;
        const int row = idxs[spl];
        const float dt = dts[spl];
#pragma unroll
        for (int ii = 0; ii < 16; ++ii) {
            const int h = ii * 32 + lane;
            const size_t o = (size_t)row * HD + h;
            const float cs = bf2f(cellA[o]), cb = bf2f(cbarA[o]);
            const float gd = bf2f(gdA[o]),   go_ = bf2f(goA[o]);
            const float cm = cb + (cs - cb) * __expf(-gd * dt);
            hs[wv][si][h] = f2bf(go_ * tanhf(cm));
        }
    }
    __syncthreads();

    // phase 2: lambda = softplus(h_samp @ W_lambda^T), summed over k<50
    float lsum[8];
#pragma unroll
    for (int i = 0; i < 8; ++i) lsum[i] = 0.f;
    const unsigned short* abase = &hs[wv][l16][0];
    for (int nt = 0; nt < 4; ++nt) {
        const unsigned short* wb = Wlam + (size_t)(nt * 16 + l16) * HD;
        v8f acc = {};
        v16bf a0 = load_frag(abase, half);
        v16bf b0 = load_frag(wb, half);
        for (int kk = 0; kk < HD; kk += 64) {
            v16bf a1 = load_frag(abase + kk + 32, half);
            v16bf b1 = load_frag(wb + kk + 32, half);
            acc = wmma_bf16(a0, b0, acc);
            if (kk + 64 < HD) {
                a0 = load_frag(abase + kk + 64, half);
                b0 = load_frag(wb + kk + 64, half);
            }
            acc = wmma_bf16(a1, b1, acc);
        }
        const int kcol = nt * 16 + l16;
        const bool valid = kcol < KK;
#pragma unroll
        for (int i = 0; i < 8; ++i) lsum[i] += valid ? splus(acc[i]) : 0.f;
    }
    // reduce over the 16 lanes of each half (N dimension), then accumulate
#pragma unroll
    for (int i = 0; i < 8; ++i) {
        float v = lsum[i];
        for (int m = 1; m < 16; m <<= 1) v += __shfl_xor(v, m, 32);
        if (!l16) {
            const int spl = sb + i + 8 * half;
            const float mk = msk[spl];
            const int bm = spl >> 8;                   // S = 256
            atomicAdd(&num[bm], v * mk);
            atomicAdd(&den[bm], mk);
        }
    }
}

// ---------------- final reduction ----------------
__global__ void k_final(const float* __restrict__ llt, const int* __restrict__ event,
                        const float* __restrict__ mask_unobs, const float* __restrict__ num,
                        const float* __restrict__ den, const float* __restrict__ duration,
                        float* __restrict__ out) {
    const int bm = threadIdx.x;
    float ac = 0.f, au = 0.f;
    for (int t = 0; t < TS; ++t) {
        const float l = llt[bm * TS + t];
        const int traw = event[bm * T2 + t + 1];
        const float mc = traw < KK ? 1.f : 0.f;
        const float mu = mask_unobs[bm * T2 + t + 1];
        ac += l * mc;
        au += l * mu;
    }
    const float integral = num[bm] / den[bm] * duration[bm];
    out[bm]      = ac - integral;
    out[RR + bm] = au - integral;
}

// ---------------- host launcher ----------------
extern "C" void kernel_launch(void* const* d_in, const int* in_sizes, int n_in,
                              void* d_out, int out_size, void* d_ws, size_t ws_size,
                              hipStream_t stream) {
    const int*   event    = (const int*)d_in[0];
    const float* dtime    = (const float*)d_in[1];
    const float* duration = (const float*)d_in[2];
    const float* dts      = (const float*)d_in[3];
    const int*   idxs     = (const int*)d_in[4];
    const float* msk      = (const float*)d_in[5];
    const float* munobs   = (const float*)d_in[6];
    const float* Emb      = (const float*)d_in[7];
    const float* W_rnn    = (const float*)d_in[8];
    const float* b_rnn    = (const float*)d_in[9];
    const float* W_lambda = (const float*)d_in[10];

    char* ws = (char*)d_ws;
    float*          P      = (float*)(ws + OFF_P);
    unsigned short* WrT    = (unsigned short*)(ws + OFF_WRT);
    unsigned short* Wlam   = (unsigned short*)(ws + OFF_WLAM);
    unsigned short* hidden = (unsigned short*)(ws + OFF_HID);
    unsigned short* cellA  = (unsigned short*)(ws + OFF_CEL);
    unsigned short* cbarA  = (unsigned short*)(ws + OFF_CBA);
    unsigned short* gdA    = (unsigned short*)(ws + OFF_GD);
    unsigned short* goA    = (unsigned short*)(ws + OFF_GO);
    float*          c_st   = (float*)(ws + OFF_CST);
    float*          cb_st  = (float*)(ws + OFF_CBS);
    float*          llt    = (float*)(ws + OFF_LLT);
    float*          num    = (float*)(ws + OFF_NUM);
    float*          den    = (float*)(ws + OFF_DEN);
    unsigned*       ctr    = (unsigned*)(ws + OFF_CTR);

    k_prep_p<<<(NE * NG + 255) / 256, 256, 0, stream>>>(Emb, W_rnn, b_rnn, P);
    k_prep_w<<<(NG * HD + 255) / 256, 256, 0, stream>>>(W_rnn, WrT);
    k_prep_misc<<<(RR * HD + 255) / 256, 256, 0, stream>>>(W_lambda, Wlam, c_st, cb_st,
                                                           num, den, ctr);
    k_recurrence<<<NWG, 512, 0, stream>>>(P, WrT, hidden, cellA, cbarA, gdA, goA,
                                          c_st, cb_st, event, dtime, ctr);
    k_llt<<<(RR * TS) / 8, 256, 0, stream>>>(hidden, W_lambda, event, llt);
    k_sample<<<(RR * SS) / (SWV * 16), SWV * 32, 0, stream>>>(cellA, cbarA, gdA, goA,
                                                              Wlam, idxs, dts, msk,
                                                              num, den);
    k_final<<<1, RR, 0, stream>>>(llt, event, munobs, num, den, duration, (float*)d_out);
}
